// GRANDBlock_1022202216983
// MI455X (gfx1250) — matrix operand order repaired and checked by reading, not compiled, not run on hardware
//
#include <hip/hip_runtime.h>
#include <hip/hip_bf16.h>

// ---------------------------------------------------------------------------
// GRAND block for MI455X (gfx1250, wave32, WMMA bf16 + TDM async tensor load).
//   B=16, C=1024, F=512, H=8, D=128, HD=1024  (qkv rows = 16384)
// All matmuls run through v_wmma_f32_16x16x32_bf16 (f32 accumulate).
// Attention V tiles are DMA'd into LDS by the Tensor Data Mover with
// hardware padding for a conflict-free LDS stride.
// ---------------------------------------------------------------------------

typedef __attribute__((ext_vector_type(16))) __bf16 v16bf;
typedef __attribute__((ext_vector_type(8)))  __bf16 v8bf;
typedef __attribute__((ext_vector_type(8)))  float  v8f;
typedef __attribute__((ext_vector_type(4)))  float  v4f;
typedef unsigned int u32x4 __attribute__((ext_vector_type(4)));
typedef int          i32x8 __attribute__((ext_vector_type(8)));
typedef int          i32x4 __attribute__((ext_vector_type(4)));

#define NROWS   16384   // B*C
#define FDIM    512
#define HDDIM   1024
#define DDIM    128
#define NB      16
#define NH      8
#define VT_STRIDE 136   // LDS stride (elems) for V^T tile: 256B row + 16B pad

#if !__has_builtin(__builtin_amdgcn_tensor_load_to_lds)
#warning "CDNA5 probe: __builtin_amdgcn_tensor_load_to_lds NOT available -> cooperative-copy fallback in use"
#endif
#if !__has_builtin(__builtin_amdgcn_s_wait_tensorcnt)
#warning "CDNA5 probe: __builtin_amdgcn_s_wait_tensorcnt NOT available -> inline asm fallback in use"
#endif

static __device__ __forceinline__ v16bf cat8(v8bf lo, v8bf hi) {
  union { v16bf v; v8bf h[2]; } u;
  u.h[0] = lo; u.h[1] = hi;
  return u.v;
}

static __device__ __forceinline__ v8f wmma_bf16(v16bf a, v16bf b, v8f c) {
  // D = A(16x32 bf16) * B(32x16 bf16) + C(16x16 f32)
  return __builtin_amdgcn_wmma_f32_16x16x32_bf16(
      /*neg_a=*/false, a, /*neg_b=*/false, b,
      /*c_mod=*/(short)0, c, /*reuse_a=*/false, /*reuse_b=*/false);
}

// convert 8 contiguous f32 -> 8 bf16 (two b128 loads)
static __device__ __forceinline__ v8bf cvt8(const float* __restrict__ p) {
  v4f a = *(const v4f*)(p);
  v4f b = *(const v4f*)(p + 4);
  v8bf r;
  r[0] = (__bf16)a[0]; r[1] = (__bf16)a[1]; r[2] = (__bf16)a[2]; r[3] = (__bf16)a[3];
  r[4] = (__bf16)b[0]; r[5] = (__bf16)b[1]; r[6] = (__bf16)b[2]; r[7] = (__bf16)b[3];
  return r;
}

// ---------------------------------------------------------------------------
// Kernel 0: transpose + fp32->bf16 convert of the weights so that WMMA
// B-fragments become two contiguous 8-element loads per lane.
//   Wk/Wq/Wv [512,1024] -> WT [1024,512];  Ww0 [1024,128] -> W0T [128,1024]
// ---------------------------------------------------------------------------
__global__ __launch_bounds__(256) void prep_weights(
    const float* __restrict__ Wk, const float* __restrict__ Wq,
    const float* __restrict__ Wv, const float* __restrict__ Ww0,
    __bf16* __restrict__ wkT, __bf16* __restrict__ wqT,
    __bf16* __restrict__ wvT, __bf16* __restrict__ w0T)
{
  const int idx = blockIdx.x * 256 + threadIdx.x;
  const int WSZ = FDIM * HDDIM;                  // 524288
  if (idx < 3 * WSZ) {
    const int m = idx / WSZ;
    const int e = idx - m * WSZ;
    const int k = e >> 10, n = e & (HDDIM - 1);  // e = k*1024 + n
    const float* W = (m == 0) ? Wk : ((m == 1) ? Wq : Wv);
    __bf16*      T = (m == 0) ? wkT : ((m == 1) ? wqT : wvT);
    T[(size_t)n * FDIM + k] = (__bf16)W[e];
  } else {
    const int e = idx - 3 * WSZ;
    if (e < HDDIM * DDIM) {
      const int k = e >> 7, n = e & (DDIM - 1);  // e = k*128 + n
      w0T[(size_t)n * HDDIM + k] = (__bf16)Ww0[e];
    }
  }
}

// ---------------------------------------------------------------------------
// Kernel 1: projection GEMM  out = x @ W + b   (one of k/q/v per blockIdx.z)
//   M=16384, N=1024, K=512. WG tile 128x128, 8 waves, wave tile 32x64.
//   A converted f32->bf16 in flight; B from pre-transposed weights.
//   k/q written flat (the reshape-without-permute head view is a memory
//   no-op); V is written PRE-TRANSPOSED per head: vbT[b,h2][d][c2], so the
//   attention P@V B-operand becomes a contiguous 2D tile (TDM-friendly).
// ---------------------------------------------------------------------------
__global__ __launch_bounds__(256) void qkv_gemm(
    const float* __restrict__ x,
    const __bf16* __restrict__ wkT, const __bf16* __restrict__ wqT,
    const __bf16* __restrict__ wvT,
    const float* __restrict__ bk, const float* __restrict__ bq,
    const float* __restrict__ bv,
    __bf16* __restrict__ kb, __bf16* __restrict__ qb, __bf16* __restrict__ vbT)
{
  const int mat = blockIdx.z;
  const __bf16* WT   = (mat == 0) ? wkT : ((mat == 1) ? wqT : wvT);
  const float*  bias = (mat == 0) ? bk  : ((mat == 1) ? bq  : bv);
  __bf16*       outb = (mat == 0) ? kb  : qb;   // used only for mat 0/1

  const int w  = threadIdx.x >> 5;
  const int l  = threadIdx.x & 31;
  const int lh = l >> 4, ll = l & 15;
  const int mrow0 = blockIdx.x * 128 + (w & 3) * 32;
  const int ncol0 = blockIdx.y * 128 + (w >> 2) * 64;

  v8f acc[2][4];
  #pragma unroll
  for (int mi = 0; mi < 2; ++mi)
    #pragma unroll
    for (int ni = 0; ni < 4; ++ni)
      acc[mi][ni] = (v8f)0.0f;

  for (int kk = 0; kk < FDIM; kk += 32) {
    v16bf a[2];
    #pragma unroll
    for (int mi = 0; mi < 2; ++mi) {
      const float* ap = x + (size_t)(mrow0 + mi * 16 + ll) * FDIM + kk + lh * 8;
      a[mi] = cat8(cvt8(ap), cvt8(ap + 16));
    }
    v16bf bfr[4];
    #pragma unroll
    for (int ni = 0; ni < 4; ++ni) {
      const __bf16* bp = WT + (size_t)(ncol0 + ni * 16 + ll) * FDIM + kk + lh * 8;
      bfr[ni] = cat8(*(const v8bf*)bp, *(const v8bf*)(bp + 16));
    }
    #pragma unroll
    for (int mi = 0; mi < 2; ++mi)
      #pragma unroll
      for (int ni = 0; ni < 4; ++ni)
        acc[mi][ni] = wmma_bf16(a[mi], bfr[ni], acc[mi][ni]);
  }

  if (mat != 2) {
    #pragma unroll
    for (int mi = 0; mi < 2; ++mi)
      #pragma unroll
      for (int ni = 0; ni < 4; ++ni) {
        const int col = ncol0 + ni * 16 + ll;
        const float bb = bias[col];
        __bf16* op = outb + (size_t)(mrow0 + mi * 16 + lh * 8) * HDDIM + col;
        #pragma unroll
        for (int i = 0; i < 8; ++i)
          op[(size_t)i * HDDIM] = (__bf16)(acc[mi][ni][i] + bb);
      }
  } else {
    // V: scatter into per-head transposed layout vbT[(b*8+h2)*128 + d][c2].
    // Within an 8-row accumulator column, c = c0 + i never crosses a
    // 128-boundary (c0 is a multiple of 8), so h2 is constant and the
    // address is affine: base + 8*i  -> static-offset stores.
    #pragma unroll
    for (int mi = 0; mi < 2; ++mi)
      #pragma unroll
      for (int ni = 0; ni < 4; ++ni) {
        const int col = ncol0 + ni * 16 + ll;   // = h*128 + dd
        const float bb = bias[col];
        const int h  = col >> 7, dd = col & 127;
        const int r0 = mrow0 + mi * 16 + lh * 8;   // = b*1024 + c0
        const int bidx = r0 >> 10, c0 = r0 & 1023;
        const int t0 = c0 * 8 + h;                 // head-view row (i = 0)
        __bf16* vp = vbT + ((size_t)(bidx * NH + (t0 >> 10)) * DDIM + dd) * 1024
                         + (t0 & 1023);
        #pragma unroll
        for (int i = 0; i < 8; ++i)
          vp[8 * i] = (__bf16)(acc[mi][ni][i] + bb);
      }
  }
}

// ---------------------------------------------------------------------------
// Kernel 2: flash attention per (b, h2); out = softmax(Q K^T / sqrt(D)) V - V
// (the "- V" fuses the (softmax - I) @ V of the reference).
//   Each WG: 128 query rows, 8 waves x 16 rows; loop over 8 KV blocks of 128.
//   V^T tile (contiguous in vbT) is DMA'd into LDS by the Tensor Data Mover
//   with hardware padding (16B per 256B row -> stride 136 elems, lanes hit
//   banks 4*lane: conflict-free). P bounced C-layout -> A-layout via 1KB/wave
//   LDS, 32 columns at a time.
// ---------------------------------------------------------------------------
__global__ __launch_bounds__(256) void attn_flash(
    const __bf16* __restrict__ qb, const __bf16* __restrict__ kb,
    const __bf16* __restrict__ vbT, __bf16* __restrict__ valsb)
{
  __shared__ __align__(16) __bf16 ldsVt[128 * VT_STRIDE];  // [d][c'] padded
  __shared__ __align__(16) __bf16 ldsP[8][16 * 32];        // per-wave P bounce

  const int w  = threadIdx.x >> 5;
  const int l  = threadIdx.x & 31;
  const int lh = l >> 4, ll = l & 15;
  const int h2 = blockIdx.y, b = blockIdx.z;
  const size_t base = (size_t)(b * NH + h2) * 1024 * DDIM;
  const int rbase = blockIdx.x * 128 + w * 16;

  const __bf16* Q  = qb + base;
  const __bf16* K  = kb + base;
  const __bf16* VT = vbT + (size_t)(b * NH + h2) * DDIM * 1024;  // [d][c2]

  // Q fragments for this wave's 16 rows (held in registers all along)
  v16bf qf[4];
  #pragma unroll
  for (int kk = 0; kk < 4; ++kk) {
    const __bf16* qp = Q + (size_t)(rbase + ll) * DDIM + kk * 32 + lh * 8;
    qf[kk] = cat8(*(const v8bf*)qp, *(const v8bf*)(qp + 16));
  }

  float mrun[8], lrun[8];
  v8f o[8];
  #pragma unroll
  for (int i = 0; i < 8; ++i) { mrun[i] = -3.0e38f; lrun[i] = 0.0f; }
  #pragma unroll
  for (int j = 0; j < 8; ++j) o[j] = (v8f)0.0f;

  const float sscale = 0.08838834764831845f;  // 1/sqrt(128)

  for (int cb = 0; cb < 8; ++cb) {
    // ---- stage V^T tile [128 d][128 c'] into LDS ----------------------
#if __has_builtin(__builtin_amdgcn_tensor_load_to_lds)
    if (w == 0) {  // one wave issues the DMA; TDM ignores EXEC
      const unsigned int lds_off =
          (unsigned int)(unsigned long long)(uintptr_t)&ldsVt[0];
      const unsigned long long ga =
          (unsigned long long)(uintptr_t)(VT + cb * 128);
      // D# group0: count=1 | lds_addr | global_addr | type=2 ("image")
      u32x4 g0 = { 1u,
                   lds_off,
                   (unsigned int)(ga & 0xffffffffu),
                   (unsigned int)((ga >> 32) & 0x1ffffffu) | (2u << 30) };
      // D# group1: data_size=1(2B), pad_enable, pad_interval=5 (256B),
      //            pad_amount=3 (16B); dims/tile/stride per CDNA5 ISA
      i32x8 g1 = { (int)((1u << 16) | (1u << 20) | (5u << 22) | (3u << 25)),
                   (int)(1024u << 16),          // tensor_dim0 = 1024 (lo16@48)
                   (int)(128u << 16),           // dim0 hi=0 | tensor_dim1 = 128
                   (int)(128u << 16),           // dim1 hi=0 | tile_dim0 = 128
                   (int)128,                    // tile_dim1 = 128, tile_dim2=0
                   (int)1024,                   // tensor_dim0_stride lo = 1024
                   0, 0 };
      i32x4 gz = { 0, 0, 0, 0 };
#if defined(__clang_major__) && (__clang_major__ >= 23)
      i32x8 gz8 = { 0, 0, 0, 0, 0, 0, 0, 0 };
      __builtin_amdgcn_tensor_load_to_lds(g0, g1, gz, gz, gz8, 0);
#else
      __builtin_amdgcn_tensor_load_to_lds(g0, g1, gz, gz, 0);
#endif
#if __has_builtin(__builtin_amdgcn_s_wait_tensorcnt)
      __builtin_amdgcn_s_wait_tensorcnt(0);
#else
      asm volatile("s_wait_tensorcnt 0x0" ::: "memory");
#endif
    }
#else
    // fallback: cooperative contiguous copy (vbT is already transposed)
    {
      const int d    = threadIdx.x >> 1;
      const int hsel = threadIdx.x & 1;
      const __bf16* vp = VT + (size_t)d * 1024 + cb * 128 + hsel * 64;
      __bf16* dp = &ldsVt[d * VT_STRIDE + hsel * 64];
      #pragma unroll
      for (int g = 0; g < 8; ++g)
        *(v8bf*)(dp + g * 8) = *(const v8bf*)(vp + g * 8);
    }
#endif
    __syncthreads();

    if (cb < 7)  // emits global_prefetch_b8 for next K tile
      __builtin_prefetch(K + (size_t)((cb + 1) * 128 + (threadIdx.x >> 1)) * DDIM, 0, 1);

    // ---- S = Q K^T for this wave's 16 rows x 128 cols -----------------
    v8f s[8];
    #pragma unroll
    for (int j = 0; j < 8; ++j) s[j] = (v8f)0.0f;
    #pragma unroll
    for (int j = 0; j < 8; ++j) {
      #pragma unroll
      for (int kk = 0; kk < 4; ++kk) {
        const __bf16* kp = K + (size_t)(cb * 128 + j * 16 + ll) * DDIM + kk * 32 + lh * 8;
        v16bf kf = cat8(*(const v8bf*)kp, *(const v8bf*)(kp + 16));
        s[j] = wmma_bf16(qf[kk], kf, s[j]);
      }
    }

    // ---- online softmax (rows live across lanes 0-15 / 16-31) --------
    // 1/sqrt(D) is folded into the exp fma (max commutes with pos. scale)
    #pragma unroll
    for (int i = 0; i < 8; ++i) {
      float tm = -3.0e38f;
      #pragma unroll
      for (int j = 0; j < 8; ++j) tm = fmaxf(tm, s[j][i]);
      tm = fmaxf(tm, __shfl_xor(tm, 1, 32));
      tm = fmaxf(tm, __shfl_xor(tm, 2, 32));
      tm = fmaxf(tm, __shfl_xor(tm, 4, 32));
      tm = fmaxf(tm, __shfl_xor(tm, 8, 32));
      const float mnew = fmaxf(mrun[i], tm * sscale);
      const float alpha = __expf(mrun[i] - mnew);
      mrun[i] = mnew;
      lrun[i] *= alpha;
      #pragma unroll
      for (int j = 0; j < 8; ++j) o[j][i] *= alpha;
      float ps = 0.0f;
      #pragma unroll
      for (int j = 0; j < 8; ++j) {
        const float p = __expf(__builtin_fmaf(s[j][i], sscale, -mnew));
        s[j][i] = p;
        ps += p;
      }
      ps += __shfl_xor(ps, 1, 32);
      ps += __shfl_xor(ps, 2, 32);
      ps += __shfl_xor(ps, 4, 32);
      ps += __shfl_xor(ps, 8, 32);
      lrun[i] += ps;
    }

    // ---- O += P @ V, 32 P-columns per chunk via LDS bounce ------------
    __bf16* Pw = &ldsP[w][0];
    #pragma unroll
    for (int kk2 = 0; kk2 < 4; ++kk2) {
      #pragma unroll
      for (int jj = 0; jj < 2; ++jj) {
        const int j = kk2 * 2 + jj;
        #pragma unroll
        for (int i = 0; i < 8; ++i)
          Pw[(lh * 8 + i) * 32 + jj * 16 + ll] = (__bf16)s[j][i];
      }
      const __bf16* pp = Pw + ll * 32 + lh * 8;
      v16bf pf = cat8(*(const v8bf*)pp, *(const v8bf*)(pp + 16));
      #pragma unroll
      for (int j = 0; j < 8; ++j) {
        const __bf16* vfp = &ldsVt[(j * 16 + ll) * VT_STRIDE + kk2 * 32 + lh * 8];
        v16bf vf = cat8(*(const v8bf*)vfp, *(const v8bf*)(vfp + 16));
        o[j] = wmma_bf16(pf, vf, o[j]);
      }
    }
    __syncthreads();
  }

  // ---- epilogue: normalize, subtract V row (the -I term), store bf16 --
  // VT[c][r] is contiguous in r -> one v8bf load per j covers all 8 rows.
  float inv[8];
  #pragma unroll
  for (int i = 0; i < 8; ++i) inv[i] = 1.0f / lrun[i];
  #pragma unroll
  for (int j = 0; j < 8; ++j) {
    const int c = j * 16 + ll;
    v8bf vrow = *(const v8bf*)(VT + (size_t)c * 1024 + rbase + lh * 8);
    __bf16* op = valsb + base + (size_t)(rbase + lh * 8) * DDIM + c;
    #pragma unroll
    for (int i = 0; i < 8; ++i) {
      const float val = __builtin_fmaf(o[j][i], inv[i], -(float)vrow[i]);
      op[(size_t)i * DDIM] = (__bf16)val;
    }
  }
}

// ---------------------------------------------------------------------------
// Kernel 3: output GEMM  out = vals_permuted @ Ww0 + bw0   (f32 out)
//   The reference's transpose(0,2,1,3) is folded into A addressing:
//   A[b*1024+c2][h2*128+d] = valsb[((b*8+h2)*1024 + c2)*128 + d]
//   M=16384, N=128, K=1024. One WG per 128 rows, wave tile 16x128.
// ---------------------------------------------------------------------------
__global__ __launch_bounds__(256) void out_gemm(
    const __bf16* __restrict__ valsb, const __bf16* __restrict__ w0T,
    const float* __restrict__ bw0, float* __restrict__ out)
{
  const int w  = threadIdx.x >> 5;
  const int l  = threadIdx.x & 31;
  const int lh = l >> 4, ll = l & 15;
  const int rbase = blockIdx.x * 128 + w * 16;
  const int m  = rbase + ll;          // A-frag row for this lane
  const int bi = m >> 10, c2 = m & 1023;

  v8f acc[8];
  #pragma unroll
  for (int j = 0; j < 8; ++j) acc[j] = (v8f)0.0f;

  #pragma unroll 4
  for (int kk = 0; kk < HDDIM; kk += 32) {
    const int h2 = kk >> 7, koff = kk & 127;
    const __bf16* ap = valsb + ((size_t)(bi * NH + h2) * 1024 + c2) * DDIM + koff + lh * 8;
    v16bf af = cat8(*(const v8bf*)ap, *(const v8bf*)(ap + 16));
    #pragma unroll
    for (int j = 0; j < 8; ++j) {
      const __bf16* bp = w0T + (size_t)(j * 16 + ll) * HDDIM + kk + lh * 8;
      v16bf bfv = cat8(*(const v8bf*)bp, *(const v8bf*)(bp + 16));
      acc[j] = wmma_bf16(af, bfv, acc[j]);
    }
  }

  #pragma unroll
  for (int j = 0; j < 8; ++j) {
    const int c = j * 16 + ll;
    const float bb = bw0[c];
    float* op = out + (size_t)(rbase + lh * 8) * DDIM + c;
    #pragma unroll
    for (int i = 0; i < 8; ++i)
      op[(size_t)i * DDIM] = acc[j][i] + bb;
  }
}

// ---------------------------------------------------------------------------
extern "C" void kernel_launch(void* const* d_in, const int* in_sizes, int n_in,
                              void* d_out, int out_size, void* d_ws, size_t ws_size,
                              hipStream_t stream)
{
  (void)in_sizes; (void)n_in; (void)out_size; (void)ws_size;
  const float* x   = (const float*)d_in[0];
  const float* Wk  = (const float*)d_in[1];
  const float* bk  = (const float*)d_in[2];
  const float* Wq  = (const float*)d_in[3];
  const float* bq  = (const float*)d_in[4];
  const float* Wv  = (const float*)d_in[5];
  const float* bv  = (const float*)d_in[6];
  const float* Ww0 = (const float*)d_in[7];
  const float* bw0 = (const float*)d_in[8];
  float* out = (float*)d_out;

  // workspace layout (bf16 buffers), offsets in bytes
  char* ws = (char*)d_ws;
  const size_t SZ_PROJ = (size_t)NROWS * HDDIM * sizeof(__bf16);  // 32 MB each
  __bf16* qb   = (__bf16*)(ws + 0 * SZ_PROJ);
  __bf16* kb   = (__bf16*)(ws + 1 * SZ_PROJ);
  __bf16* vbT  = (__bf16*)(ws + 2 * SZ_PROJ);   // per-head transposed V
  __bf16* vals = (__bf16*)(ws + 3 * SZ_PROJ);
  char* wsw = ws + 4 * SZ_PROJ;
  __bf16* wkT = (__bf16*)(wsw);
  __bf16* wqT = (__bf16*)(wsw + 1 * (size_t)FDIM * HDDIM * sizeof(__bf16));
  __bf16* wvT = (__bf16*)(wsw + 2 * (size_t)FDIM * HDDIM * sizeof(__bf16));
  __bf16* w0T = (__bf16*)(wsw + 3 * (size_t)FDIM * HDDIM * sizeof(__bf16));

  // 0) weight transpose + convert
  {
    const int total = 3 * FDIM * HDDIM + HDDIM * DDIM;
    prep_weights<<<dim3((total + 255) / 256), dim3(256), 0, stream>>>(
        Wk, Wq, Wv, Ww0, wkT, wqT, wvT, w0T);
  }
  // 1) projections k/q/v (z selects matrix; V lands pre-transposed per head)
  qkv_gemm<<<dim3(NROWS / 128, HDDIM / 128, 3), dim3(256), 0, stream>>>(
      x, wkT, wqT, wvT, bk, bq, bv, kb, qb, vbT);
  // 2) flash attention per (row-tile, h2, b)
  attn_flash<<<dim3(8, NH, NB), dim3(256), 0, stream>>>(qb, kb, vbT, vals);
  // 3) output projection
  out_gemm<<<dim3(NROWS / 128), dim3(256), 0, stream>>>(vals, w0T, bw0, out);
}
